// AxialAttention_19404662243634
// MI455X (gfx1250) — matrix operand order repaired and checked by reading, not compiled, not run on hardware
//
#include <hip/hip_runtime.h>
#include <hip/hip_bf16.h>

typedef __attribute__((ext_vector_type(8)))  __bf16 v8bf;
typedef __attribute__((ext_vector_type(16))) __bf16 v16bf;
typedef __attribute__((ext_vector_type(8)))  float  v8f;
typedef int v4i_gcc __attribute__((vector_size(16)));   // matches builtin param type

union BF16Frag { v16bf v; v8bf h[2]; };

// ---- CDNA5 async global->LDS DMA (guarded; fallback = reg double-buffer) ----
#if defined(__gfx1250__) &&                                              \
    __has_builtin(__builtin_amdgcn_global_load_async_to_lds_b128) &&     \
    __has_builtin(__builtin_amdgcn_s_wait_asynccnt)
#define ASYNC_LDS 1
#else
#define ASYNC_LDS 0
#endif

#if ASYNC_LDS
__device__ __forceinline__ void async_cp16(const __bf16* g, __bf16* l) {
  __builtin_amdgcn_global_load_async_to_lds_b128(
      (__attribute__((address_space(1))) v4i_gcc*)(g),
      (__attribute__((address_space(3))) v4i_gcc*)(l), 0, 0);
}
__device__ __forceinline__ void async_wait0() {
  __builtin_amdgcn_s_wait_asynccnt(0);
}
#endif

// ---------------------------------------------------------------- utilities

__global__ void cvt_f32_bf16_kernel(const float* __restrict__ src,
                                    __bf16* __restrict__ dst, int n) {
  int i = blockIdx.x * blockDim.x + threadIdx.x;
  if (i < n) dst[i] = (__bf16)src[i];
}

// rel[h][i][j] = w[h][(j - i) + T - 1];  grid = NH*T blocks, block = T threads
__global__ void rel_bias_kernel(const float* __restrict__ w,
                                float* __restrict__ rel, int T) {
  int h = blockIdx.x / T, i = blockIdx.x % T, j = threadIdx.x;
  rel[((size_t)h * T + i) * T + j] = w[(size_t)h * (2 * T - 1) + (j - i) + (T - 1)];
}

// out[b][c][r] = in[b][r][c]   (batched 2-D transpose, 32x32 tiles)
__global__ void batched_transpose(const float* __restrict__ in,
                                  float* __restrict__ out, int R, int Ccol) {
  __shared__ float tile[32][33];
  int b = blockIdx.z;
  int r0 = blockIdx.y * 32, c0 = blockIdx.x * 32;
  const float* inb = in + (size_t)b * R * Ccol;
  float* outb = out + (size_t)b * R * Ccol;
  int tx = threadIdx.x, ty = threadIdx.y;   // 32 x 8
  for (int j = 0; j < 32; j += 8)
    tile[ty + j][tx] = inb[(size_t)(r0 + ty + j) * Ccol + c0 + tx];
  __syncthreads();
  for (int j = 0; j < 32; j += 8)
    outb[(size_t)(c0 + ty + j) * R + r0 + tx] = tile[tx][ty + j];
}

// LayerNorm over C=384; one wave per token; optionally two gain/bias outputs.
__global__ __launch_bounds__(256)
void ln_dual_kernel(const float* __restrict__ x,
                    const float* __restrict__ g1, const float* __restrict__ b1,
                    __bf16* __restrict__ o1,
                    const float* __restrict__ g2, const float* __restrict__ b2,
                    __bf16* __restrict__ o2) {
  int wid = threadIdx.x >> 5, lane = threadIdx.x & 31;
  int t = blockIdx.x * 8 + wid;
  const float* xr = x + (size_t)t * 384;
  float v[12];
  float s = 0.f;
  for (int i = 0; i < 12; i++) { v[i] = xr[lane + i * 32]; s += v[i]; }
  for (int o = 16; o; o >>= 1) s += __shfl_xor(s, o);
  float mean = s * (1.f / 384.f);
  float q = 0.f;
  for (int i = 0; i < 12; i++) { float d = v[i] - mean; q += d * d; }
  for (int o = 16; o; o >>= 1) q += __shfl_xor(q, o);
  float inv = rsqrtf(q * (1.f / 384.f) + 1e-4f);
  for (int i = 0; i < 12; i++) {
    int c = lane + i * 32;
    float nv = (v[i] - mean) * inv;
    o1[(size_t)t * 384 + c] = (__bf16)(nv * g1[c] + b1[c]);
    if (o2) o2[(size_t)t * 384 + c] = (__bf16)(nv * g2[c] + b2[c]);
  }
}

// ---------------------------------------------------------------- WMMA GEMM
// C[M,N] = A[M,K] @ B[N,K]^T + bias   (bf16 in, f32 accum)
// Block tile 128x128, 8 waves (4x2) of 32x64; K-step 32, double-buffered LDS.
// Async path: tile k+1 DMA'd to LDS (ASYNCcnt) while WMMAs consume tile k.
template <bool ACCUM, bool GELU, bool WRITE_H>
__global__ __launch_bounds__(256)
void gemm_bf16_wmma(const __bf16* __restrict__ A, const __bf16* __restrict__ B,
                    const float* __restrict__ bias,
                    float* __restrict__ outF, __bf16* __restrict__ outH,
                    int M, int N, int K) {
  __shared__ __bf16 As[2][128][40];   // row stride 80B (16B-aligned)
  __shared__ __bf16 Bs[2][128][40];
  const int tid  = threadIdx.x;
  const int lane = tid & 31, wv = tid >> 5;
  const int wm = wv >> 1, wn = wv & 1;                 // 4 x 2 wave grid
  const int bm = blockIdx.y * 128, bn = blockIdx.x * 128;
  const int half = lane >> 4, l16 = lane & 15;

  // staging: 128x32 bf16 per matrix = 512 16B-chunks; thread t owns chunk t
  // (row r0, 8-col group c0) and chunk t+256 (row r0+64, same col group).
  const int r0 = tid >> 2, c0 = (tid & 3) * 8;
  const __bf16* Ar0 = A + (size_t)(bm + r0) * K + c0;
  const __bf16* Ar1 = A + (size_t)(bm + r0 + 64) * K + c0;
  const __bf16* Br0 = B + (size_t)(bn + r0) * K + c0;
  const __bf16* Br1 = B + (size_t)(bn + r0 + 64) * K + c0;

#if ASYNC_LDS
  auto stage = [&](int buf, int k0) {
    async_cp16(Ar0 + k0, &As[buf][r0][c0]);
    async_cp16(Ar1 + k0, &As[buf][r0 + 64][c0]);
    async_cp16(Br0 + k0, &Bs[buf][r0][c0]);
    async_cp16(Br1 + k0, &Bs[buf][r0 + 64][c0]);
  };
#else
  v8bf pf[4];
  auto gload = [&](int k0) {
    pf[0] = *(const v8bf*)(Ar0 + k0);
    pf[1] = *(const v8bf*)(Ar1 + k0);
    pf[2] = *(const v8bf*)(Br0 + k0);
    pf[3] = *(const v8bf*)(Br1 + k0);
  };
  auto commit = [&](int buf) {
    *(v8bf*)&As[buf][r0][c0]      = pf[0];
    *(v8bf*)&As[buf][r0 + 64][c0] = pf[1];
    *(v8bf*)&Bs[buf][r0][c0]      = pf[2];
    *(v8bf*)&Bs[buf][r0 + 64][c0] = pf[3];
  };
#endif

  v8f acc[2][4] = {};

#if ASYNC_LDS
  stage(0, 0);
  async_wait0();
#else
  gload(0);
  commit(0);
#endif
  __syncthreads();

  int buf = 0;
  for (int k0 = 0; k0 < K; k0 += 32) {
    const bool has_next = (k0 + 32 < K);
#if ASYNC_LDS
    if (has_next) stage(buf ^ 1, k0 + 32);     // DMA next tile during compute
#else
    if (has_next) gload(k0 + 32);              // issue global loads early
#endif

    BF16Frag a[2], b[4];
    for (int tm = 0; tm < 2; tm++) {
      int mr = wm * 32 + tm * 16 + l16;
      a[tm].h[0] = *(const v8bf*)&As[buf][mr][half * 8];
      a[tm].h[1] = *(const v8bf*)&As[buf][mr][16 + half * 8];
    }
    for (int tn = 0; tn < 4; tn++) {
      int nc = wn * 64 + tn * 16 + l16;
      b[tn].h[0] = *(const v8bf*)&Bs[buf][nc][half * 8];
      b[tn].h[1] = *(const v8bf*)&Bs[buf][nc][16 + half * 8];
    }
    for (int tm = 0; tm < 2; tm++)
      for (int tn = 0; tn < 4; tn++)
        acc[tm][tn] = __builtin_amdgcn_wmma_f32_16x16x32_bf16(
            false, a[tm].v, false, b[tn].v, (short)0, acc[tm][tn], false, false);

#if ASYNC_LDS
    async_wait0();
#else
    if (has_next) commit(buf ^ 1);
#endif
    __syncthreads();
    buf ^= 1;
  }

  for (int tm = 0; tm < 2; tm++)
    for (int tn = 0; tn < 4; tn++) {
      int col = bn + wn * 64 + tn * 16 + l16;
      float bv = bias ? bias[col] : 0.f;
      int mbase = bm + wm * 32 + tm * 16 + half * 8;  // C layout: VGPR r -> M r / r+8
      for (int r = 0; r < 8; r++) {
        size_t idx = (size_t)(mbase + r) * N + col;
        float vres = acc[tm][tn][r] + bv;
        if (ACCUM) vres += outF[idx];
        if (GELU)  vres = vres * 0.5f * (1.f + erff(vres * 0.70710678f));
        if (ACCUM) outF[idx] = vres;
        if (WRITE_H) outH[idx] = (__bf16)vres;
      }
    }
}

// ---------------------------------------------------------------- attention
// One block = one (sequence, head): T=64, hd=64. 4 waves, each owns 16 rows.
// S = Q K^T / 8 + rel -> softmax -> Y = P V, all matmuls via WMMA bf16.
__global__ __launch_bounds__(128)
void axial_attn_kernel(const __bf16* __restrict__ qkv,  // [tok][1152]
                       __bf16* __restrict__ yout,       // [tok][384]
                       const float* __restrict__ rel,   // [6][64][64]
                       int stride) {                    // 64 = row attn, 1 = col
  __shared__ __bf16 Vt[64][72];  // V transposed: Vt[d][s]
  __shared__ __bf16 Pl[64][72];  // softmax probs, bf16
  const int tid = threadIdx.x;
  const int lane = tid & 31, warp = tid >> 5;
  const int half = lane >> 4, l16 = lane & 15;
  const int head = blockIdx.x % 6;
  const int seq  = blockIdx.x / 6;
  const int base = (stride == 64) ? (seq >> 6) * 4096 + (seq & 63)
                                  : (seq >> 6) * 4096 + (seq & 63) * 64;
  const int hc = head * 64;

  // stage V transposed
  for (int idx = tid; idx < 4096; idx += 128) {
    int sp = idx >> 6, d = idx & 63;
    Vt[d][sp] = qkv[(size_t)(base + sp * stride) * 1152 + 768 + hc + d];
  }
  __syncthreads();

  // Q fragments for this wave's 16 rows (K = 0..63 -> two 16x32 frags)
  BF16Frag aq[2];
  {
    int mrow = warp * 16 + l16;
    const __bf16* qp = qkv + (size_t)(base + mrow * stride) * 1152 + hc;
    for (int kb = 0; kb < 2; kb++) {
      aq[kb].h[0] = *(const v8bf*)&qp[kb * 32 + half * 8];
      aq[kb].h[1] = *(const v8bf*)&qp[kb * 32 + 16 + half * 8];
    }
  }

  // S = Q K^T  (4 column tiles of 16)
  v8f sc[4];
  for (int jt = 0; jt < 4; jt++) {
    v8f accs = {};
    int scol = jt * 16 + l16;
    const __bf16* kp = qkv + (size_t)(base + scol * stride) * 1152 + 384 + hc;
    for (int kb = 0; kb < 2; kb++) {
      BF16Frag bk;
      bk.h[0] = *(const v8bf*)&kp[kb * 32 + half * 8];
      bk.h[1] = *(const v8bf*)&kp[kb * 32 + 16 + half * 8];
      accs = __builtin_amdgcn_wmma_f32_16x16x32_bf16(
          false, aq[kb].v, false, bk.v, (short)0, accs, false, false);
    }
    sc[jt] = accs;
  }

  // softmax per row (row = warp*16 + half*8 + r, cols split over 16 lanes x 4 tiles)
  const float* relh = rel + head * 4096;
  for (int r = 0; r < 8; r++) {
    int ti = warp * 16 + half * 8 + r;
    float m = -1e30f;
    for (int jt = 0; jt < 4; jt++) {
      int tj = jt * 16 + l16;
      sc[jt][r] = sc[jt][r] * 0.125f + relh[ti * 64 + tj];
      m = fmaxf(m, sc[jt][r]);
    }
    for (int o = 8; o; o >>= 1) m = fmaxf(m, __shfl_xor(m, o, 16));
    float sum = 0.f;
    for (int jt = 0; jt < 4; jt++) { sc[jt][r] = __expf(sc[jt][r] - m); sum += sc[jt][r]; }
    for (int o = 8; o; o >>= 1) sum += __shfl_xor(sum, o, 16);
    float rinv = 1.f / sum;
    for (int jt = 0; jt < 4; jt++)
      Pl[ti][jt * 16 + l16] = (__bf16)(sc[jt][r] * rinv);
  }

  // Y = P V   (P rows owned by this wave; Vt provides K-contiguous columns)
  for (int jt = 0; jt < 4; jt++) {
    v8f accy = {};
    for (int kb = 0; kb < 2; kb++) {
      BF16Frag ap, bv;
      int mrow = warp * 16 + l16;
      ap.h[0] = *(const v8bf*)&Pl[mrow][kb * 32 + half * 8];
      ap.h[1] = *(const v8bf*)&Pl[mrow][kb * 32 + 16 + half * 8];
      int d = jt * 16 + l16;
      bv.h[0] = *(const v8bf*)&Vt[d][kb * 32 + half * 8];
      bv.h[1] = *(const v8bf*)&Vt[d][kb * 32 + 16 + half * 8];
      accy = __builtin_amdgcn_wmma_f32_16x16x32_bf16(
          false, ap.v, false, bv.v, (short)0, accy, false, false);
    }
    for (int r = 0; r < 8; r++) {
      int m = warp * 16 + half * 8 + r;
      yout[(size_t)(base + m * stride) * 384 + hc + jt * 16 + l16] = (__bf16)accy[r];
    }
  }
}

// ---------------------------------------------------------------- launcher

extern "C" void kernel_launch(void* const* d_in, const int* in_sizes, int n_in,
                              void* d_out, int out_size, void* d_ws, size_t ws_size,
                              hipStream_t stream) {
  (void)in_sizes; (void)n_in; (void)out_size; (void)ws_size;
  const float* x       = (const float*)d_in[0];
  const float* rln1_g  = (const float*)d_in[1];
  const float* rln1_b  = (const float*)d_in[2];
  const float* cln1_g  = (const float*)d_in[3];
  const float* cln1_b  = (const float*)d_in[4];
  const float* ln2_g   = (const float*)d_in[5];
  const float* ln2_b   = (const float*)d_in[6];
  const float* row_q_w = (const float*)d_in[7];
  const float* row_q_b = (const float*)d_in[8];
  const float* row_k_w = (const float*)d_in[9];
  const float* row_k_b = (const float*)d_in[10];
  const float* row_v_w = (const float*)d_in[11];
  const float* row_v_b = (const float*)d_in[12];
  const float* row_p_w = (const float*)d_in[13];
  const float* row_p_b = (const float*)d_in[14];
  const float* col_q_w = (const float*)d_in[15];
  const float* col_q_b = (const float*)d_in[16];
  const float* col_k_w = (const float*)d_in[17];
  const float* col_k_b = (const float*)d_in[18];
  const float* col_v_w = (const float*)d_in[19];
  const float* col_v_b = (const float*)d_in[20];
  const float* col_p_w = (const float*)d_in[21];
  const float* col_p_b = (const float*)d_in[22];
  const float* ff_w1   = (const float*)d_in[23];
  const float* ff_b1   = (const float*)d_in[24];
  const float* ff_w2   = (const float*)d_in[25];
  const float* ff_b2   = (const float*)d_in[26];
  const float* row_bias_w = (const float*)d_in[27];
  const float* col_bias_w = (const float*)d_in[28];

  // ---- workspace layout (bytes, 256-aligned) ----
  char* ws = (char*)d_ws;
  const size_t off_yacc  = 0;                         // 32768*384 f32
  const size_t off_lnrow = 50331648;                  // 32768*384 bf16 (reused for LN2 out)
  const size_t off_lncol = 75497472;                  // 32768*384 bf16
  const size_t off_big   = 100663296;                 // qkv(72MB)+yattn(24MB) | hbuf(96MB)
  const size_t off_w     = 201326592;                 // bf16 weights
  const size_t off_bias  = off_w + 4718592;           // fused qkv biases f32
  const size_t off_rel   = off_bias + 9216;           // rel biases f32

  float*  yacc  = (float*)(ws + off_yacc);
  __bf16* lnrow = (__bf16*)(ws + off_lnrow);
  __bf16* lncol = (__bf16*)(ws + off_lncol);
  __bf16* qkvb  = (__bf16*)(ws + off_big);
  __bf16* yattn = (__bf16*)(ws + off_big + 75497472);
  __bf16* hbuf  = (__bf16*)(ws + off_big);            // overlaps qkv/yattn (disjoint in time)
  __bf16* wbase = (__bf16*)(ws + off_w);
  __bf16* wqkv_row = wbase;
  __bf16* wqkv_col = wbase + 442368;
  __bf16* wp_row   = wbase + 884736;
  __bf16* wp_col   = wbase + 1032192;
  __bf16* w1       = wbase + 1179648;
  __bf16* w2       = wbase + 1769472;
  float* bqkv_row = (float*)(ws + off_bias);
  float* bqkv_col = bqkv_row + 1152;
  float* rel_r = (float*)(ws + off_rel);
  float* rel_c = rel_r + 24576;

  auto cvt = [&](const float* src, __bf16* dst, int n) {
    cvt_f32_bf16_kernel<<<(n + 255) / 256, 256, 0, stream>>>(src, dst, n);
  };
  // fused QKV weights [1152,384] + projection + FF weights -> bf16
  cvt(row_q_w, wqkv_row,          147456);
  cvt(row_k_w, wqkv_row + 147456, 147456);
  cvt(row_v_w, wqkv_row + 294912, 147456);
  cvt(col_q_w, wqkv_col,          147456);
  cvt(col_k_w, wqkv_col + 147456, 147456);
  cvt(col_v_w, wqkv_col + 294912, 147456);
  cvt(row_p_w, wp_row, 147456);
  cvt(col_p_w, wp_col, 147456);
  cvt(ff_w1, w1, 589824);
  cvt(ff_w2, w2, 589824);
  // fused QKV biases (f32, d2d)
  (void)hipMemcpyAsync(bqkv_row,       row_q_b, 1536, hipMemcpyDeviceToDevice, stream);
  (void)hipMemcpyAsync(bqkv_row + 384, row_k_b, 1536, hipMemcpyDeviceToDevice, stream);
  (void)hipMemcpyAsync(bqkv_row + 768, row_v_b, 1536, hipMemcpyDeviceToDevice, stream);
  (void)hipMemcpyAsync(bqkv_col,       col_q_b, 1536, hipMemcpyDeviceToDevice, stream);
  (void)hipMemcpyAsync(bqkv_col + 384, col_k_b, 1536, hipMemcpyDeviceToDevice, stream);
  (void)hipMemcpyAsync(bqkv_col + 768, col_v_b, 1536, hipMemcpyDeviceToDevice, stream);
  // relative position biases
  rel_bias_kernel<<<6 * 64, 64, 0, stream>>>(row_bias_w, rel_r, 64);
  rel_bias_kernel<<<6 * 64, 64, 0, stream>>>(col_bias_w, rel_c, 64);

  // x [8,384,4096] -> y_acc [8,4096,384] (residual base x0, f32)
  batched_transpose<<<dim3(128, 12, 8), dim3(32, 8), 0, stream>>>(x, yacc, 384, 4096);
  // LN (row + col params in one pass)
  ln_dual_kernel<<<4096, 256, 0, stream>>>(yacc, rln1_g, rln1_b, lnrow,
                                           cln1_g, cln1_b, lncol);

  const dim3 blk(256);
  // ---- row attention ----
  gemm_bf16_wmma<false, false, true><<<dim3(9, 256), blk, 0, stream>>>(
      lnrow, wqkv_row, bqkv_row, nullptr, qkvb, 32768, 1152, 384);
  axial_attn_kernel<<<512 * 6, 128, 0, stream>>>(qkvb, yattn, rel_r, 64);
  gemm_bf16_wmma<true, false, false><<<dim3(3, 256), blk, 0, stream>>>(
      yattn, wp_row, row_p_b, yacc, nullptr, 32768, 384, 384);
  // ---- col attention ----
  gemm_bf16_wmma<false, false, true><<<dim3(9, 256), blk, 0, stream>>>(
      lncol, wqkv_col, bqkv_col, nullptr, qkvb, 32768, 1152, 384);
  axial_attn_kernel<<<512 * 6, 128, 0, stream>>>(qkvb, yattn, rel_c, 1);
  gemm_bf16_wmma<true, false, false><<<dim3(3, 256), blk, 0, stream>>>(
      yattn, wp_col, col_p_b, yacc, nullptr, 32768, 384, 384);
  // ---- feed-forward ----
  ln_dual_kernel<<<4096, 256, 0, stream>>>(yacc, ln2_g, ln2_b, lnrow,
                                           nullptr, nullptr, nullptr);
  gemm_bf16_wmma<false, true, true><<<dim3(12, 256), blk, 0, stream>>>(
      lnrow, w1, ff_b1, nullptr, hbuf, 32768, 1536, 384);
  gemm_bf16_wmma<true, false, false><<<dim3(3, 256), blk, 0, stream>>>(
      hbuf, w2, ff_b2, yacc, nullptr, 32768, 384, 1536);
  // y_acc [8,4096,384] -> out [8,384,4096]
  batched_transpose<<<dim3(12, 128, 8), dim3(32, 8), 0, stream>>>(
      yacc, (float*)d_out, 4096, 384);
}
